// OctreeConvBnRelu_63651415327311
// MI455X (gfx1250) — matrix operand order repaired and checked by reading, not compiled, not run on hardware
//
#include <hip/hip_runtime.h>
#include <hip/hip_bf16.h>
#include <stdint.h>

// ---------------------------------------------------------------------------
// OctreeConvBnRelu for gfx1250 (MI455X, wave32, WMMA).
//   conv  : out(64 x N) = Wt(64 x 864) * col(864 x N), col gathered via neigh
//   K-dim : K = k*32 + c  -> each neighbor gather row == one K=32 bf16 slice
//   GEMM  : 1 wave = 16 points x 64 channels (4 acc tiles, 4 WMMA per gather)
//           weights staged in LDS via async global->LDS (ASYNCcnt) copies
//           indices preloaded (1-level gather chain), 3-deep B prefetch
//           branchless gather via zero-row sentinel (EXEC stays all-ones)
//   BN    : deterministic 2-stage reduction + fused scale/bias/relu
// ---------------------------------------------------------------------------

typedef __attribute__((ext_vector_type(16))) __bf16 v16bf;
typedef __attribute__((ext_vector_type(8)))  __bf16 v8bf;
typedef __attribute__((ext_vector_type(8)))  float  v8f;

#define C_IN    32
#define C_OUT   64
#define KNB     27              // neighbors per point
#define KDIM    (C_IN * KNB)    // 864
#define WPITCH  872             // LDS row pitch in ushorts (52 dwords mod 64 banks)
#define LDS_BYTES (C_OUT * WPITCH * 2)   // 111,616 B dynamic LDS
#define RCHUNKS 100             // partial-reduction chunks per channel

__device__ __forceinline__ unsigned short f32_to_bf16_rne(float f) {
  uint32_t u = __float_as_uint(f);
  uint32_t r = u + 0x7FFFu + ((u >> 16) & 1u);   // round-to-nearest-even
  return (unsigned short)(r >> 16);
}

// ---- prep 1: x (C_IN x N, f32) -> xb (N+1 x C_IN, bf16); row N = zeros -----
__global__ void xpose_bf16_kernel(const float* __restrict__ x,
                                  unsigned short* __restrict__ xb, int N) {
  int n = blockIdx.x * blockDim.x + threadIdx.x;
  if (n > N) return;
  unsigned short row[C_IN];
  if (n < N) {
#pragma unroll
    for (int c = 0; c < C_IN; ++c)
      row[c] = f32_to_bf16_rne(x[(size_t)c * N + n]);  // coalesced per-c reads
  } else {
#pragma unroll
    for (int c = 0; c < C_IN; ++c) row[c] = 0;         // sentinel zero row
  }
  uint4* dst = reinterpret_cast<uint4*>(xb + (size_t)n * C_IN);
  const uint4* src = reinterpret_cast<const uint4*>(row);
  dst[0] = src[0]; dst[1] = src[1]; dst[2] = src[2]; dst[3] = src[3];
}

// ---- prep 2: weight[o][c][k] (f32) -> Wt[o][k*32+c] (bf16) -----------------
__global__ void wprep_kernel(const float* __restrict__ w,
                             unsigned short* __restrict__ wt) {
  int t = blockIdx.x * blockDim.x + threadIdx.x;
  if (t >= C_OUT * KDIM) return;
  int o  = t / KDIM;
  int j  = t % KDIM;
  int kk = j >> 5;
  int c  = j & 31;
  wt[t] = f32_to_bf16_rne(w[(size_t)o * KDIM + c * KNB + kk]);
}

// ---- main gather-GEMM: 8 waves/block, wave = 16 points x 64 channels -------
__global__ __launch_bounds__(256)
void conv_wmma_kernel(const unsigned short* __restrict__ xb,
                      const unsigned short* __restrict__ wt,
                      const long long* __restrict__ neigh,
                      float* __restrict__ out, int N, int ngroups) {
  extern __shared__ unsigned short wls[];   // [C_OUT][WPITCH]

  const int lane = threadIdx.x & 31;
  const int wave = threadIdx.x >> 5;
  const int grp  = blockIdx.x * 8 + wave;   // 16-point group id
  const bool active = (grp < ngroups);      // wave-uniform

  const int nbase = grp * 16;
  const int nloc  = lane & 15;              // tile row (A) / tile col (B,D)
  const int hi    = lane >> 4;              // half-wave select

  // ---- preload all 27 neighbor indices (overlaps with async LDS fill) -----
  int ei[KNB];
  if (active) {
    const long long* nrow = neigh + (size_t)(nbase + nloc) * KNB;
#pragma unroll
    for (int kk = 0; kk < KNB; ++kk) {
      long long idx = nrow[kk];
      ei[kk] = (idx < 0) ? N : (int)idx;    // sentinel zero row, branchless
    }
  }

  // ---- async weight stage: 64 rows x 1728 B, 4 threads/row, 27 x 16B each --
  {
    const int row = threadIdx.x >> 2;       // 0..63
    const int c0  = threadIdx.x & 3;        // 0..3
    unsigned goff = (unsigned)row * (KDIM * 2) + (unsigned)c0 * 16;
    unsigned loff = (unsigned)(uintptr_t)&wls[row * WPITCH] + (unsigned)c0 * 16;
#pragma unroll
    for (int j = 0; j < 27; ++j) {
      unsigned l = loff + (unsigned)j * 64;
      unsigned g = goff + (unsigned)j * 64;
      asm volatile("global_load_async_to_lds_b128 %0, %1, %2"
                   :: "v"(l), "v"(g), "s"(wt)
                   : "memory");
    }
    asm volatile("s_wait_asynccnt 0x0" ::: "memory");
  }
  __syncthreads();
  if (!active) return;                      // wave-uniform: EXEC stays full

  // B gather: lane holds col nloc, K = hi*16 .. hi*16+15 (contiguous c slice)
  auto loadB = [&](int kk) -> v16bf {
    return *reinterpret_cast<const v16bf*>(xb + (size_t)ei[kk] * C_IN + hi * 16);
  };
  // A from LDS: row M = m*16 + nloc; lane<16 holds K 0..15, lane>=16 K 16..31
  const unsigned short* arow = &wls[nloc * WPITCH + hi * 16];

  v8f acc0 = {}, acc1 = {}, acc2 = {}, acc3 = {};
  v16bf b0 = loadB(0), b1 = loadB(1), b2 = loadB(2);   // 3-deep prefetch

#pragma unroll
  for (int kk = 0; kk < KNB; ++kk) {
    v16bf b = b0;
    b0 = b1; b1 = b2;
    if (kk + 3 < KNB) b2 = loadB(kk + 3);

    v16bf a0, a1, a2, a3;
#pragma unroll
    for (int m = 0; m < 4; ++m) {
      const unsigned short* ap = arow + m * 16 * WPITCH + kk * 32;
      v8bf lo = *reinterpret_cast<const v8bf*>(ap);
      v8bf hh = *reinterpret_cast<const v8bf*>(ap + 8);
      v16bf a;
#pragma unroll
      for (int i = 0; i < 8; ++i) { a[i] = lo[i]; a[8 + i] = hh[i]; }
      if (m == 0) a0 = a; else if (m == 1) a1 = a;
      else if (m == 2) a2 = a; else a3 = a;
    }
    // 4 independent WMMAs sharing B: no D->A/B hazard, XDL stays busy
    acc0 = __builtin_amdgcn_wmma_f32_16x16x32_bf16(false, a0, false, b, (short)0, acc0, false, false);
    acc1 = __builtin_amdgcn_wmma_f32_16x16x32_bf16(false, a1, false, b, (short)0, acc1, false, false);
    acc2 = __builtin_amdgcn_wmma_f32_16x16x32_bf16(false, a2, false, b, (short)0, acc2, false, false);
    acc3 = __builtin_amdgcn_wmma_f32_16x16x32_bf16(false, a3, false, b, (short)0, acc3, false, false);
  }

  // D layout: VGPR v -> row v (lanes 0-15) / row v+8 (lanes 16-31)
  const int col = nbase + nloc;
#pragma unroll
  for (int v = 0; v < 8; ++v) {
    int r = v + hi * 8;
    out[(size_t)(r)      * N + col] = acc0[v];
    out[(size_t)(r + 16) * N + col] = acc1[v];
    out[(size_t)(r + 32) * N + col] = acc2[v];
    out[(size_t)(r + 48) * N + col] = acc3[v];
  }
}

// ---- scalar tail (only if N % 16 != 0; exact f32 path) ---------------------
__global__ void conv_tail_kernel(const float* __restrict__ x,
                                 const float* __restrict__ w,
                                 const long long* __restrict__ neigh,
                                 float* __restrict__ out, int N, int n0) {
  int n = n0 + blockIdx.x;
  int o = threadIdx.x;
  if (n >= N || o >= C_OUT) return;
  float acc = 0.f;
  for (int kk = 0; kk < KNB; ++kk) {
    long long idx = neigh[(size_t)n * KNB + kk];
    if (idx < 0) continue;
    for (int c = 0; c < C_IN; ++c)
      acc += w[(size_t)o * KDIM + c * KNB + kk] * x[(size_t)c * N + idx];
  }
  out[(size_t)o * N + n] = acc;
}

// ---- BN stage 1: per-(channel, chunk) sum / sumsq (deterministic) ----------
__global__ void reduce_kernel(const float* __restrict__ out,
                              float* __restrict__ part, int N) {
  __shared__ float ssum[256], ssq[256];
  const int ch = blockIdx.y;
  const int cb = blockIdx.x;
  const int chunk = (N + RCHUNKS - 1) / RCHUNKS;
  const int start = cb * chunk;
  const int end   = start + chunk < N ? start + chunk : N;
  const float* p = out + (size_t)ch * N;
  float s = 0.f, q = 0.f;
  for (int i = start + threadIdx.x; i < end; i += 256) {
    float v = p[i];
    s += v; q += v * v;
  }
  ssum[threadIdx.x] = s; ssq[threadIdx.x] = q;
  __syncthreads();
  for (int off = 128; off > 0; off >>= 1) {
    if ((int)threadIdx.x < off) {
      ssum[threadIdx.x] += ssum[threadIdx.x + off];
      ssq[threadIdx.x]  += ssq[threadIdx.x + off];
    }
    __syncthreads();
  }
  if (threadIdx.x == 0) {
    part[((size_t)ch * RCHUNKS + cb) * 2 + 0] = ssum[0];
    part[((size_t)ch * RCHUNKS + cb) * 2 + 1] = ssq[0];
  }
}

// ---- BN stage 2: fold partials -> per-channel scale/bias -------------------
__global__ void finalize_kernel(const float* __restrict__ part,
                                const float* __restrict__ gamma,
                                const float* __restrict__ beta,
                                float* __restrict__ sb, int N) {
  __shared__ float ssum[128], ssq[128];
  const int ch = blockIdx.x;
  float s = 0.f, q = 0.f;
  if ((int)threadIdx.x < RCHUNKS) {
    s = part[((size_t)ch * RCHUNKS + threadIdx.x) * 2 + 0];
    q = part[((size_t)ch * RCHUNKS + threadIdx.x) * 2 + 1];
  }
  ssum[threadIdx.x] = s; ssq[threadIdx.x] = q;
  __syncthreads();
  for (int off = 64; off > 0; off >>= 1) {
    if ((int)threadIdx.x < off) {
      ssum[threadIdx.x] += ssum[threadIdx.x + off];
      ssq[threadIdx.x]  += ssq[threadIdx.x + off];
    }
    __syncthreads();
  }
  if (threadIdx.x == 0) {
    float invN  = 1.0f / (float)N;
    float mean  = ssum[0] * invN;
    float var   = fmaxf(ssq[0] * invN - mean * mean, 0.f);
    float scale = gamma[ch] * rsqrtf(var + 1e-5f);
    sb[ch]          = scale;
    sb[C_OUT + ch]  = beta[ch] - mean * scale;
  }
}

// ---- BN apply + ReLU, in place on d_out ------------------------------------
__global__ void apply_kernel(float* __restrict__ out,
                             const float* __restrict__ sb, int N) {
  int n  = blockIdx.x * blockDim.x + threadIdx.x;
  int ch = blockIdx.y;
  if (n >= N) return;
  size_t i = (size_t)ch * N + n;
  out[i] = fmaxf(fmaf(out[i], sb[ch], sb[C_OUT + ch]), 0.f);
}

// ---------------------------------------------------------------------------
extern "C" void kernel_launch(void* const* d_in, const int* in_sizes, int n_in,
                              void* d_out, int out_size, void* d_ws, size_t ws_size,
                              hipStream_t stream) {
  const float*     x     = (const float*)d_in[0];      // (32, N) after squeeze
  const long long* neigh = (const long long*)d_in[1];  // (N, 27) int64
  const float*     w     = (const float*)d_in[2];      // (64, 32, 27)
  const float*     gamma = (const float*)d_in[3];
  const float*     beta  = (const float*)d_in[4];
  float*           out   = (float*)d_out;              // (64, N)

  const int N = in_sizes[1] / KNB;

  // workspace carve-up (~13 MB total)
  char* ws = (char*)d_ws;
  size_t off = 0;
  unsigned short* xb = (unsigned short*)(ws + off);    // N+1 rows (zero row)
  off += (size_t)(N + 1) * C_IN * sizeof(unsigned short); off = (off + 255) & ~(size_t)255;
  unsigned short* wt = (unsigned short*)(ws + off);
  off += (size_t)C_OUT * KDIM * sizeof(unsigned short);   off = (off + 255) & ~(size_t)255;
  float* part = (float*)(ws + off);
  off += (size_t)C_OUT * RCHUNKS * 2 * sizeof(float);     off = (off + 255) & ~(size_t)255;
  float* sb = (float*)(ws + off);

  xpose_bf16_kernel<<<(N + 1 + 255) / 256, 256, 0, stream>>>(x, xb, N);
  wprep_kernel<<<(C_OUT * KDIM + 255) / 256, 256, 0, stream>>>(w, wt);

  const int ngroups = N / 16;                 // full 16-point groups
  if (ngroups > 0) {
    const int nblocks = (ngroups + 7) / 8;    // 8 waves per block
    conv_wmma_kernel<<<nblocks, 256, LDS_BYTES, stream>>>(xb, wt, neigh, out,
                                                          N, ngroups);
  }
  const int ntail = N - ngroups * 16;
  if (ntail > 0)
    conv_tail_kernel<<<ntail, C_OUT, 0, stream>>>(x, w, neigh, out, N,
                                                  ngroups * 16);

  dim3 rgrid(RCHUNKS, C_OUT);
  reduce_kernel<<<rgrid, 256, 0, stream>>>(out, part, N);
  finalize_kernel<<<C_OUT, 128, 0, stream>>>(part, gamma, beta, sb, N);
  dim3 agrid((N + 255) / 256, C_OUT);
  apply_kernel<<<agrid, 256, 0, stream>>>(out, sb, N);
}
